// LSTM_SelfAttention_Model_54288386622090
// MI455X (gfx1250) — matrix operand order repaired and checked by reading, not compiled, not run on hardware
//
#include <hip/hip_runtime.h>

// ---------------------------------------------------------------------------
// Model dims
// ---------------------------------------------------------------------------
#define Bdim   64
#define Sdim   128
#define Fdim   16
#define Hdim   64
#define G4H    256    // 4*H
#define QKVdim 192    // 3*H
#define FRAG   512    // halfs per 16x32 / 32x16 fragment (32 lanes x 16)

typedef _Float16 half_t;
typedef __attribute__((ext_vector_type(16))) _Float16 v16h;
typedef __attribute__((ext_vector_type(8)))  float    v8f;

// ---------------------------------------------------------------------------
// WMMA + swizzled-fragment helpers (wave32 layouts per CDNA5 ISA 7.12.2)
// ---------------------------------------------------------------------------
__device__ __forceinline__ v8f wmma_f16(v16h a, v16h b, v8f c) {
  return __builtin_amdgcn_wmma_f32_16x16x32_f16(
      false, a, false, b, (short)0, c, false, false);
}

// Position of element (row, k) inside a lane-swizzled A fragment (16x32 f16).
__device__ __forceinline__ int a_swz(int row, int k) {
  const int lane = row + (((k >> 3) & 1) << 4);
  const int i    = (k & 7) + ((k >> 4) << 3);
  return lane * 16 + i;
}

// Position of element (k, col) inside a lane-swizzled B fragment (32x16 f16).
__device__ __forceinline__ int b_swz(int k, int col) {
  const int lane = col + (((k >> 3) & 1) << 4);
  return lane * 16 + (k & 7) + ((k >> 4) << 3);
}

// Fragment load: this lane's 16 contiguous halfs (32B, 32B-aligned) -> 2x b128.
__device__ __forceinline__ v16h load_frag(const half_t* __restrict__ base) {
  const int lane = threadIdx.x & 31;
  return *(const v16h*)(base + lane * 16);
}

__device__ __forceinline__ float sigmoidf_(float x) {
  return 1.0f / (1.0f + __expf(-x));
}

__device__ __forceinline__ float fast_tanhf(float x) {
#if __has_builtin(__builtin_amdgcn_tanhf)
  return __builtin_amdgcn_tanhf(x);
#elif __has_builtin(__builtin_amdgcn_tanh_f32)
  return __builtin_amdgcn_tanh_f32(x);
#else
  return tanhf(x);
#endif
}

// ---------------------------------------------------------------------------
// Kernel 1: batched per-feature LSTM, fully register-resident cell update.
// One block per feature. Wave w owns (rt = w>>1, hTiles {2(w&1), 2(w&1)+1}).
// For each cell it computes all four gate tiles (cols ht, ht+4, ht+8, ht+12),
// so {i,f,g,o} for a given (b,h) land in the same lane/element -> no gate LDS,
// c state in registers. h ping-pongs in LDS (A-frag order) -> 1 barrier/step.
// ---------------------------------------------------------------------------
__global__ __launch_bounds__(256)
void lstm_kernel(const float* __restrict__ x,     // [B,S,F]
                 const float* __restrict__ Wih,   // [F,4H]
                 const float* __restrict__ Whh,   // [F,4H,H]
                 const float* __restrict__ bih,   // [F,4H]
                 const float* __restrict__ bhh,   // [F,4H]
                 half_t* __restrict__ out_sw)     // [B,F,16 frags,512] A-swz
{
  const int f    = blockIdx.x;
  const int tid  = threadIdx.x;
  const int wave = tid >> 5;
  const int lane = tid & 31;

  extern __shared__ char smem[];
  half_t* sWhh  = (half_t*)(smem);               // 32 frags x 512  32 KB (B-swz)
  half_t* sH    = (half_t*)(smem + 32 * 1024);   // 2 x 8 frags     16 KB (A-swz, ping-pong)
  float*  sWihL = (float*)(smem + 48 * 1024);    // [256]
  float*  sBias = sWihL + G4H;                   // [256]
  float*  sX    = sBias + G4H;                   // [2][64]

  // ---- init: swizzle+convert recurrent weights, biases, zero state, x(0)
  const float* Whhf = Whh + (size_t)f * G4H * Hdim;
  for (int idx = tid; idx < G4H * Hdim; idx += 256) {
    const int g = idx >> 6, h = idx & 63;     // Whh[g][h]; B[k=h][n=g]
    sWhh[(((h >> 5) << 4) + (g >> 4)) * FRAG + b_swz(h & 31, g & 15)] =
        (half_t)Whhf[idx];
  }
  if (tid < G4H) {
    sWihL[tid] = Wih[f * G4H + tid];
    sBias[tid] = bih[f * G4H + tid] + bhh[f * G4H + tid];
  }
  for (int idx = tid; idx < 2 * 8 * FRAG; idx += 256) sH[idx] = (half_t)0.0f;
  if (tid < Bdim) sX[tid] = x[((size_t)tid * Sdim + 0) * Fdim + f];
  __syncthreads();

  const int rt      = wave >> 1;                 // batch-row tile 0..3
  const int ht0     = (wave & 1) << 1;           // first h tile of this wave
  const int col16   = lane & 15;
  const int rowBase = rt * 16 + ((lane < 16) ? 0 : 8);

  // hoisted per-lane gate constants (bias, Wih) for both cells
  float biasv[2][4], wihv[2][4];
#pragma unroll
  for (int cc = 0; cc < 2; ++cc) {
    const int h = (ht0 + cc) * 16 + col16;
#pragma unroll
    for (int gi = 0; gi < 4; ++gi) {
      biasv[cc][gi] = sBias[gi * Hdim + h];
      wihv[cc][gi]  = sWihL[gi * Hdim + h];
    }
  }
  float creg[2][8];
#pragma unroll
  for (int cc = 0; cc < 2; ++cc)
#pragma unroll
    for (int r = 0; r < 8; ++r) creg[cc][r] = 0.0f;

  for (int s = 0; s < Sdim; ++s) {
    const int pg = s & 1;
    half_t* sHr = sH + pg * (8 * FRAG);
    half_t* sHw = sH + (pg ^ 1) * (8 * FRAG);
    const float* sXr = sX + pg * Bdim;
    float* sXw = sX + (pg ^ 1) * Bdim;

    const v16h a0 = load_frag(sHr + (rt * 2 + 0) * FRAG);
    const v16h a1 = load_frag(sHr + (rt * 2 + 1) * FRAG);

#pragma unroll
    for (int cc = 0; cc < 2; ++cc) {
      const int ht = ht0 + cc;
      const int h  = ht * 16 + col16;            // this lane's h index
      v8f acc[4];
#pragma unroll
      for (int gi = 0; gi < 4; ++gi) {
        const int ctg = gi * 4 + ht;             // gate column tile
        v8f z = {};
        acc[gi] = wmma_f16(a0, load_frag(sWhh + (0 * 16 + ctg) * FRAG), z);
        acc[gi] = wmma_f16(a1, load_frag(sWhh + (1 * 16 + ctg) * FRAG), acc[gi]);
      }
      const int kfh = h >> 5;                    // k-frag of h within A layout
      const int osw = a_swz(s & 15, h & 31);     // out element offset (rows = s)
#pragma unroll
      for (int r = 0; r < 8; ++r) {
        const int row = rowBase + r;             // batch index b
        const float xw = sXr[row];
        const float gi_ = sigmoidf_(acc[0][r] + biasv[cc][0] + xw * wihv[cc][0]);
        const float gf_ = sigmoidf_(acc[1][r] + biasv[cc][1] + xw * wihv[cc][1]);
        const float gg_ = fast_tanhf(acc[2][r] + biasv[cc][2] + xw * wihv[cc][2]);
        const float go_ = sigmoidf_(acc[3][r] + biasv[cc][3] + xw * wihv[cc][3]);
        const float c   = gf_ * creg[cc][r] + gi_ * gg_;
        creg[cc][r] = c;
        const float hh = go_ * fast_tanhf(c);
        sHw[(rt * 2 + kfh) * FRAG + a_swz(row & 15, h & 31)] = (half_t)hh;
        out_sw[(((size_t)row * Fdim + f) * 16 + ((s >> 4) << 1) + kfh) * FRAG +
               osw] = (half_t)hh;
      }
    }
    if (tid < Bdim && s + 1 < Sdim)
      sXw[tid] = x[((size_t)tid * Sdim + (s + 1)) * Fdim + f];
    __syncthreads();
  }
}

// ---------------------------------------------------------------------------
// Kernel 2: per-(b,f) attention, operands in fragment-swizzled layouts.
// ---------------------------------------------------------------------------
__global__ __launch_bounds__(256)
void attn_kernel(const half_t* __restrict__ lstm_sw, // [B,F,16 frags,512] A-swz
                 const float* __restrict__ Wqkv,     // [F,3H,H]
                 const float* __restrict__ bqkv,     // [F,3H]
                 const float* __restrict__ Wout,     // [F,H,H]
                 const float* __restrict__ bout,     // [F,H]
                 float* __restrict__ attn_out,       // [F,B,S,S]
                 float* __restrict__ pooled)         // [B, F*H]
{
  const int b    = blockIdx.x;
  const int f    = blockIdx.y;
  const int tid  = threadIdx.x;
  const int wave = tid >> 5;
  const int lane = tid & 31;

  extern __shared__ char smem[];
  half_t* sQ    = (half_t*)(smem);                 // 16 frags (rt,head) 16 KB A-swz
  half_t* sK    = (half_t*)(smem + 16 * 1024);     // 16 frags (ct,head) 16 KB B-swz
  half_t* sV    = (half_t*)(smem + 32 * 1024);     // [128][64]          16 KB
  half_t* sWq   = (half_t*)(smem + 48 * 1024);     // 24 frags (ct,kf)   24 KB B-swz
  float*  sS    = (float*)(smem + 72 * 1024);      // [128][128]         64 KB
  float*  sW    = (float*)(smem + 136 * 1024);     // [128][128]         64 KB
  float*  sWbar = (float*)(smem + 200 * 1024);     // [128]
  float*  sVec  = sWbar + 128;                     // [64]

  const half_t* X  = lstm_sw + ((size_t)(b * Fdim + f)) * (16 * FRAG);
  const float*  Wf = Wqkv + (size_t)f * QKVdim * Hdim;
  const float*  bf = bqkv + f * QKVdim;
  float* dst = attn_out + ((size_t)(f * Bdim + b)) * (Sdim * Sdim);

  // ---- stage QKV weights into LDS in B-fragment order (once per block)
  for (int idx = tid; idx < QKVdim * Hdim; idx += 256) {
    const int g = idx >> 6, h = idx & 63;     // W[g][h]; B[k=h][n=g]
    sWq[(((g >> 4) << 1) + (h >> 5)) * FRAG + b_swz(h & 31, g & 15)] =
        (half_t)Wf[idx];
  }
  __syncthreads();

  // ---- QKV projection: [128x192] = X @ W^T, 8x12 tiles, K=64
  for (int t = wave; t < 96; t += 8) {
    const int rt = t / 12, ct = t % 12;
    v8f acc = {};
#pragma unroll
    for (int kf = 0; kf < 2; ++kf) {
      v16h a  = load_frag(X + ((rt << 1) + kf) * FRAG);
      v16h bb = load_frag(sWq + ((ct << 1) + kf) * FRAG);
      acc = wmma_f16(a, bb, acc);
    }
    const int g       = ct * 16 + (lane & 15);
    const int rowBase = rt * 16 + ((lane < 16) ? 0 : 8);
    const float bias  = bf[g];
#pragma unroll
    for (int r = 0; r < 8; ++r) {
      const int srow  = rowBase + r;
      const float val = acc[r] + bias;
      if (g < 64) {
        // q, pre-scaled by 1/sqrt(hd); A-frag order, frag = (rt, head)
        sQ[(((srow >> 4) << 1) + (g >> 5)) * FRAG + a_swz(srow & 15, g & 31)] =
            (half_t)(val * 0.17677669529663687f);
      } else if (g < 128) {
        // kT; B-frag order, frag = (key tile, head)
        const int d = g - 64;
        sK[(((srow >> 4) << 1) + (d >> 5)) * FRAG + b_swz(d & 31, srow & 15)] =
            (half_t)val;
      } else {
        sV[srow * 64 + (g - 128)] = (half_t)val;
      }
    }
  }
  __syncthreads();

  for (int n = 0; n < 2; ++n) {   // heads
    // ---- scores[128x128] = q_head @ kT_head (K=32 -> one WMMA per tile)
    for (int t = wave; t < 64; t += 8) {
      const int rt = t >> 3, ct = t & 7;
      v16h a  = load_frag(sQ + ((rt << 1) + n) * FRAG);
      v16h bb = load_frag(sK + ((ct << 1) + n) * FRAG);
      v8f acc = {};
      acc = wmma_f16(a, bb, acc);
      const int col     = ct * 16 + (lane & 15);
      const int rowBase = rt * 16 + ((lane < 16) ? 0 : 8);
#pragma unroll
      for (int r = 0; r < 8; ++r) sS[(rowBase + r) * 128 + col] = acc[r];
    }
    __syncthreads();

    // ---- softmax along rows (in place)
    if (tid < 128) {
      float* row = sS + tid * 128;
      float m = row[0];
      for (int k = 1; k < 128; ++k) m = fmaxf(m, row[k]);
      float sum = 0.0f;
      for (int k = 0; k < 128; ++k) { float e = __expf(row[k] - m); row[k] = e; sum += e; }
      const float inv = 1.0f / sum;
      for (int k = 0; k < 128; ++k) row[k] *= inv;
    }
    __syncthreads();

    // ---- column means (pooled path) + head-mean; head 1 streams to output
    if (tid < 128) {
      float s = 0.0f;
      for (int q = 0; q < 128; ++q) s += sS[q * 128 + tid];
      sWbar[tid] = s * (1.0f / 128.0f);
    }
    if (n == 0) {
      for (int idx = tid; idx < 128 * 128; idx += 256)
        sW[idx] = 0.5f * sS[idx];
    } else {
      for (int idx = tid; idx < 128 * 128; idx += 256)
        __builtin_nontemporal_store(sW[idx] + 0.5f * sS[idx], dst + idx);
    }
    __syncthreads();

    // ---- ovec_head = wbar @ v_head (exact mean-pool of w@v)
    if (tid < 32) {
      float acc = 0.0f;
      for (int k = 0; k < 128; ++k)
        acc += sWbar[k] * (float)sV[k * 64 + n * 32 + tid];
      sVec[n * 32 + tid] = acc;
    }
    __syncthreads();
  }

  // ---- pooled = ovec @ Wout^T + bout
  if (tid < 64) {
    const float* Wo = Wout + ((size_t)f * Hdim + tid) * Hdim;
    float acc = bout[f * Hdim + tid];
    for (int h = 0; h < Hdim; ++h) acc += sVec[h] * Wo[h];
    pooled[(size_t)b * (Fdim * Hdim) + f * Hdim + tid] = acc;
  }
}

// ---------------------------------------------------------------------------
// Kernel 3: static/time MLPs + final FC (tiny; one block)
// ---------------------------------------------------------------------------
__global__ __launch_bounds__(256)
void head_kernel(const float* __restrict__ stat,  // [64,32]
                 const float* __restrict__ tg,    // [64,1]
                 const float* __restrict__ d1_w, const float* __restrict__ d1_b,
                 const float* __restrict__ d2_w, const float* __restrict__ d2_b,
                 const float* __restrict__ t1_w, const float* __restrict__ t1_b,
                 const float* __restrict__ t2_w, const float* __restrict__ t2_b,
                 const float* __restrict__ fc_w, const float* __restrict__ fc_b,
                 const float* __restrict__ pooled_wd,
                 const float* __restrict__ pooled_rd,
                 float* __restrict__ out)          // [64,2]
{
  __shared__ float s1[64][64];
  __shared__ float s2[64][32];
  __shared__ float t1v[64][16];
  __shared__ float t2v[64][8];
  const int tid = threadIdx.x;

  for (int idx = tid; idx < 64 * 64; idx += 256) {
    const int b = idx >> 6, j = idx & 63;
    float acc = d1_b[j];
    for (int k = 0; k < 32; ++k) acc += stat[b * 32 + k] * d1_w[j * 32 + k];
    s1[b][j] = fmaxf(acc, 0.0f);
  }
  for (int idx = tid; idx < 64 * 16; idx += 256) {
    const int b = idx >> 4, j = idx & 15;
    t1v[b][j] = fmaxf(tg[b] * t1_w[j] + t1_b[j], 0.0f);
  }
  __syncthreads();
  for (int idx = tid; idx < 64 * 32; idx += 256) {
    const int b = idx >> 5, j = idx & 31;
    float acc = d2_b[j];
    for (int k = 0; k < 64; ++k) acc += s1[b][k] * d2_w[j * 64 + k];
    s2[b][j] = fmaxf(acc, 0.0f);
  }
  for (int idx = tid; idx < 64 * 8; idx += 256) {
    const int b = idx >> 3, j = idx & 7;
    float acc = t2_b[j];
    for (int k = 0; k < 16; ++k) acc += t1v[b][k] * t2_w[j * 16 + k];
    t2v[b][j] = fmaxf(acc, 0.0f);
  }
  __syncthreads();
  for (int idx = tid; idx < 128; idx += 256) {
    const int b = idx >> 1, o = idx & 1;
    const float* w  = fc_w + o * 2088;
    const float* pw = pooled_wd + (size_t)b * 1024;
    const float* pr = pooled_rd + (size_t)b * 1024;
    float acc = fc_b[o];
    for (int j = 0; j < 1024; ++j) acc += pw[j] * w[j];
    for (int j = 0; j < 1024; ++j) acc += pr[j] * w[1024 + j];
    for (int j = 0; j < 32;   ++j) acc += s2[b][j] * w[2048 + j];
    for (int j = 0; j < 8;    ++j) acc += t2v[b][j] * w[2080 + j];
    out[b * 2 + o] = acc;
  }
}

// ---------------------------------------------------------------------------
// Launch
// ---------------------------------------------------------------------------
extern "C" void kernel_launch(void* const* d_in, const int* in_sizes, int n_in,
                              void* d_out, int out_size, void* d_ws, size_t ws_size,
                              hipStream_t stream) {
  (void)in_sizes; (void)n_in; (void)out_size; (void)ws_size;

  const float* wd_x    = (const float*)d_in[0];
  const float* rd_x    = (const float*)d_in[1];
  const float* stat    = (const float*)d_in[2];
  const float* tg      = (const float*)d_in[3];
  const float* wd_Wih  = (const float*)d_in[4];
  const float* wd_Whh  = (const float*)d_in[5];
  const float* wd_bih  = (const float*)d_in[6];
  const float* wd_bhh  = (const float*)d_in[7];
  const float* rd_Wih  = (const float*)d_in[8];
  const float* rd_Whh  = (const float*)d_in[9];
  const float* rd_bih  = (const float*)d_in[10];
  const float* rd_bhh  = (const float*)d_in[11];
  const float* wd_aw   = (const float*)d_in[12];
  const float* wd_ab   = (const float*)d_in[13];
  const float* wd_ow   = (const float*)d_in[14];
  const float* wd_ob   = (const float*)d_in[15];
  const float* rd_aw   = (const float*)d_in[16];
  const float* rd_ab   = (const float*)d_in[17];
  const float* rd_ow   = (const float*)d_in[18];
  const float* rd_ob   = (const float*)d_in[19];
  const float* d1_w    = (const float*)d_in[20];
  const float* d1_b    = (const float*)d_in[21];
  const float* d2_w    = (const float*)d_in[22];
  const float* d2_b    = (const float*)d_in[23];
  const float* t1_w    = (const float*)d_in[24];
  const float* t1_b    = (const float*)d_in[25];
  const float* t2_w    = (const float*)d_in[26];
  const float* t2_b    = (const float*)d_in[27];
  const float* fc_w    = (const float*)d_in[28];
  const float* fc_b    = (const float*)d_in[29];

  float* out     = (float*)d_out;                 // [64,2]
  float* wd_attn = out + 128;                     // [16,64,128,128]
  float* rd_attn = wd_attn + (size_t)16 * 64 * 128 * 128;

  char* ws = (char*)d_ws;
  const size_t lstmBytes = (size_t)Bdim * Fdim * Sdim * Hdim * sizeof(half_t); // 16 MB
  half_t* lstm_wd   = (half_t*)(ws);
  half_t* lstm_rd   = (half_t*)(ws + lstmBytes);
  float*  pooled_wd = (float*)(ws + 2 * lstmBytes);
  float*  pooled_rd = pooled_wd + (size_t)Bdim * Fdim * Hdim;

  const size_t lds_lstm = 48 * 1024 + (G4H + G4H + 2 * Bdim) * sizeof(float); // ~50.5 KB
  const size_t lds_attn = 200 * 1024 + (128 + 64) * sizeof(float);            // ~200.8 KB

  lstm_kernel<<<Fdim, 256, lds_lstm, stream>>>(wd_x, wd_Wih, wd_Whh, wd_bih, wd_bhh, lstm_wd);
  lstm_kernel<<<Fdim, 256, lds_lstm, stream>>>(rd_x, rd_Wih, rd_Whh, rd_bih, rd_bhh, lstm_rd);

  attn_kernel<<<dim3(Bdim, Fdim), 256, lds_attn, stream>>>(
      lstm_wd, wd_aw, wd_ab, wd_ow, wd_ob, wd_attn, pooled_wd);
  attn_kernel<<<dim3(Bdim, Fdim), 256, lds_attn, stream>>>(
      lstm_rd, rd_aw, rd_ab, rd_ow, rd_ob, rd_attn, pooled_rd);

  head_kernel<<<1, 256, 0, stream>>>(stat, tg, d1_w, d1_b, d2_w, d2_b,
                                     t1_w, t1_b, t2_w, t2_b, fc_w, fc_b,
                                     pooled_wd, pooled_rd, out);
}
